// StackAugmentedRNN_9981503996215
// MI455X (gfx1250) — matrix operand order repaired and checked by reading, not compiled, not run on hardware
//
#include <hip/hip_runtime.h>
#include <stdint.h>

// ---- problem sizes ----
#define VOC 45
#define HID 1500
#define SW  50
#define SD  10
#define TT  128
#define G4H 6000            // 4*HID
#define KP  1536            // HID padded to mult of 32 (WMMA K)
#define KPS 64              // SW padded to mult of 32
#define MT  375             // G4H/16 row tiles
#define NT  8               // 128/16 col tiles
#define NPROJ (VOC + 3 + SW)   // 98 wide dot-products per step

typedef __attribute__((ext_vector_type(16))) __bf16 v16bf;
typedef __attribute__((ext_vector_type(8)))  float  v8f;

__device__ __forceinline__ unsigned short f2bf(float f) {
    unsigned int u = __float_as_uint(f);
    unsigned int r = u + 0x7FFFu + ((u >> 16) & 1u);
    return (unsigned short)(r >> 16);
}
__device__ __forceinline__ float bf2f(unsigned short u) {
    return __uint_as_float(((unsigned int)u) << 16);
}
__device__ __forceinline__ float sigf(float x) { return 1.0f / (1.0f + expf(-x)); }

__device__ __forceinline__ float waveReduce(float s) {
    #pragma unroll
    for (int off = 16; off; off >>= 1) s += __shfl_xor(s, off, 32);
    return s;
}

__device__ __forceinline__ v16bf ldfrag(const unsigned short* p0, const unsigned short* p1) {
    v16bf a;
    ((uint4*)&a)[0] = *(const uint4*)p0;
    ((uint4*)&a)[1] = *(const uint4*)p1;
    return a;
}
__device__ __forceinline__ v16bf zfrag() {
    v16bf a;
    uint4 z = {0u, 0u, 0u, 0u};
    ((uint4*)&a)[0] = z; ((uint4*)&a)[1] = z;
    return a;
}

// ---------- repack fp32 -> bf16, zero-padded K ----------
__global__ void kPack(unsigned short* __restrict__ dst, const float* __restrict__ src,
                      int rows, int srcCols, int dstCols, int srcOff, int active) {
    long n = (long)rows * dstCols;
    for (long i = (long)blockIdx.x * blockDim.x + threadIdx.x; i < n;
         i += (long)gridDim.x * blockDim.x) {
        int r = (int)(i / dstCols), c = (int)(i % dstCols);
        float v = (c < active) ? src[(long)r * srcCols + srcOff + c] : 0.0f;
        dst[i] = f2bf(v);
    }
}

// ---------- state init (runs every call: deterministic) ----------
__global__ void kInit(float* h, float* c, float* st, unsigned short* hb,
                      const float* h0, const float* c0, const float* s0) {
    int i = blockIdx.x * blockDim.x + threadIdx.x;
    if (i < 2 * HID) { h[i] = h0[i]; c[i] = c0[i]; }
    if (i < SD * SW) st[i] = s0[i];
    if (i < 2 * KP) {
        int dir = i / KP, j = i % KP;
        hb[i] = (j < HID) ? f2bf(h0[dir * HID + j]) : (unsigned short)0;
    }
}

// ---------- hoisted GEMM: Xb[dir][t][m] = W_ih[:, :H] @ emb[tok_t] + b_ih + b_hh ----------
__global__ void kXproj(const unsigned short* __restrict__ WxF, const unsigned short* __restrict__ WxB,
                       const unsigned short* __restrict__ Eb, const int* __restrict__ toks,
                       const float* __restrict__ biF, const float* __restrict__ bhF,
                       const float* __restrict__ biB, const float* __restrict__ bhB,
                       float* __restrict__ Xb) {
    int wave = (blockIdx.x * blockDim.x + threadIdx.x) >> 5;
    int lane = threadIdx.x & 31;
    if (wave >= 2 * MT * NT) return;
    int dir = wave / (MT * NT);
    int rem = wave % (MT * NT);
    int row0 = (rem / NT) * 16, t0 = (rem % NT) * 16;

    const unsigned short* W = dir ? WxB : WxF;
    int m    = row0 + (lane & 15);
    int tcol = t0 + (lane & 15);
    int tok  = toks[tcol];
    const unsigned short* arow = W + (size_t)m * KP + ((lane & 16) ? 8 : 0);
    const unsigned short* brow = Eb + (size_t)tok * KP + ((lane & 16) ? 8 : 0);

    v8f acc = {0.f, 0.f, 0.f, 0.f, 0.f, 0.f, 0.f, 0.f};
    for (int kb = 0; kb < KP; kb += 32) {
        v16bf a = ldfrag(arow + kb, arow + kb + 16);
        v16bf b = ldfrag(brow + kb, brow + kb + 16);
        acc = __builtin_amdgcn_wmma_f32_16x16x32_bf16(false, a, false, b,
                                                      (short)0, acc, false, false);
    }
    int mr = row0 + ((lane & 16) ? 8 : 0);
    const float* bi = dir ? biB : biF;
    const float* bh = dir ? bhB : bhF;
    float* out = Xb + ((size_t)dir * TT + (size_t)(t0 + (lane & 15))) * G4H;
    #pragma unroll
    for (int r = 0; r < 8; ++r)
        out[mr + r] = acc[r] + bi[mr + r] + bh[mr + r];
}

// ---------- per-step: softmax(ctrl) + stack update (tiny serial kernel) ----------
// rawCS = [ctrl_raw(3) | si_tanh(50)] computed by kProj from h(t-1)
__global__ void kStack(const float* __restrict__ rawCS, float* __restrict__ stackS,
                       unsigned short* __restrict__ stb) {
    __shared__ float sCtrl[3];
    __shared__ float sNew[SD * SW];
    int tid = threadIdx.x;
    if (tid == 0) {
        float a = rawCS[0], b = rawCS[1], cc = rawCS[2];
        float m = fmaxf(a, fmaxf(b, cc));
        float e0 = expf(a - m), e1 = expf(b - m), e2 = expf(cc - m);
        float inv = 1.0f / (e0 + e1 + e2);
        sCtrl[0] = e0 * inv; sCtrl[1] = e1 * inv; sCtrl[2] = e2 * inv;
    }
    __syncthreads();
    if (tid < SD * SW) {
        int d = tid / SW, w = tid % SW;
        float cur = stackS[tid];
        float up = (d == 0) ? rawCS[3 + w] : stackS[tid - SW];
        float dn = (d == SD - 1) ? 0.f : stackS[tid + SW];
        sNew[tid] = sCtrl[2] * cur + sCtrl[0] * up + sCtrl[1] * dn;
    }
    __syncthreads();
    if (tid < SD * SW) stackS[tid] = sNew[tid];
    if (tid < KPS) stb[tid] = (tid < SW) ? f2bf(sNew[tid]) : (unsigned short)0;
}

// ---------- per-step: gates = Whh@h + Ws@stack_top + Xb[t] ----------
// one WORKGROUP per 16-row tile; K split across the 8 waves (6 chunks each),
// LDS reduction of the column-0 partials. 750 WGs -> full-chip L2 streaming.
__global__ void kGates(const unsigned short* __restrict__ WhF, const unsigned short* __restrict__ WhB,
                       const unsigned short* __restrict__ WsF, const unsigned short* __restrict__ WsB,
                       const unsigned short* __restrict__ hb, const unsigned short* __restrict__ stb,
                       const float* __restrict__ Xb, float* __restrict__ gates, int t) {
    __shared__ float sPart[8][16];
    int wg   = blockIdx.x;                 // 0 .. 2*MT-1
    int dir  = wg / MT;
    int row0 = (wg % MT) * 16;
    int lane = threadIdx.x & 31;
    int wv   = threadIdx.x >> 5;           // 0..7

    const unsigned short* W  = dir ? WhB : WhF;
    const unsigned short* Ws = dir ? WsB : WsF;
    const unsigned short* hv = hb + (size_t)dir * KP;
    int m = row0 + (lane & 15);
    const unsigned short* arow  = W  + (size_t)m * KP  + ((lane & 16) ? 8 : 0);
    const unsigned short* arow2 = Ws + (size_t)m * KPS + ((lane & 16) ? 8 : 0);
    bool bl = (lane == 0) || (lane == 16);      // B column 0 only
    int bko = (lane & 16) ? 8 : 0;

    v8f acc = {0.f, 0.f, 0.f, 0.f, 0.f, 0.f, 0.f, 0.f};
    int kb0 = wv * 6 * 32;                  // 48 K-chunks / 8 waves
    #pragma unroll
    for (int i = 0; i < 6; ++i) {
        int kb = kb0 + i * 32;
        v16bf a = ldfrag(arow + kb, arow + kb + 16);
        v16bf b = bl ? ldfrag(hv + bko + kb, hv + bko + kb + 16) : zfrag();
        acc = __builtin_amdgcn_wmma_f32_16x16x32_bf16(false, a, false, b,
                                                      (short)0, acc, false, false);
    }
    if (wv < 2) {                           // stack slice: 2 chunks on waves 0/1
        int kb = wv * 32;
        v16bf a = ldfrag(arow2 + kb, arow2 + kb + 16);
        v16bf b = bl ? ldfrag(stb + bko + kb, stb + bko + kb + 16) : zfrag();
        acc = __builtin_amdgcn_wmma_f32_16x16x32_bf16(false, a, false, b,
                                                      (short)0, acc, false, false);
    }
    if (bl) {
        int half = lane >> 4;               // 0: rows r, 1: rows 8+r
        #pragma unroll
        for (int r = 0; r < 8; ++r) sPart[wv][half * 8 + r] = acc[r];
    }
    __syncthreads();
    if (threadIdx.x < 16) {
        float s = 0.f;
        #pragma unroll
        for (int w = 0; w < 8; ++w) s += sPart[w][threadIdx.x];
        int mr = row0 + threadIdx.x;
        gates[(size_t)dir * G4H + mr] = s + Xb[((size_t)dir * TT + t) * G4H + mr];
    }
}

// ---------- per-step: LSTM elementwise update (wide) ----------
__global__ void kUpdate(const float* __restrict__ gates, float* __restrict__ c,
                        float* __restrict__ h, unsigned short* __restrict__ hb) {
    int idx = blockIdx.x * blockDim.x + threadIdx.x;
    if (idx >= 2 * HID) return;
    int dir = idx / HID, j = idx % HID;
    const float* g = gates + (size_t)dir * G4H;
    float gi = sigf(g[j]);
    float gf = sigf(g[HID + j]);
    float gg = tanhf(g[2 * HID + j]);
    float go = sigf(g[3 * HID + j]);
    float c2 = gf * c[idx] + gi * gg;
    float h2 = go * tanhf(c2);
    c[idx] = c2; h[idx] = h2;
    hb[(size_t)dir * KP + j] = f2bf(h2);
}

// ---------- per-step: all 98 wide dot products on h(t) ----------
// waves 0..44  -> logits(t)            (decoder)
// waves 45..47 -> ctrl_raw for t+1
// waves 48..97 -> tanh(si) for t+1
__global__ void kProj(const float* __restrict__ h,
                      const unsigned short* __restrict__ WdB, const float* __restrict__ bd,
                      const unsigned short* __restrict__ WcB, const float* __restrict__ bc,
                      const unsigned short* __restrict__ WsiB, const float* __restrict__ bsi,
                      float* __restrict__ outT, float* __restrict__ rawCS) {
    int wave = (blockIdx.x * blockDim.x + threadIdx.x) >> 5;
    int lane = threadIdx.x & 31;
    if (wave >= NPROJ) return;
    const unsigned short* row;
    if (wave < VOC)          row = WdB  + (size_t)wave * 2 * HID;
    else if (wave < VOC + 3) row = WcB  + (size_t)(wave - VOC) * 2 * HID;
    else                     row = WsiB + (size_t)(wave - VOC - 3) * 2 * HID;
    float s = 0.f;
    for (int k = lane; k < 2 * HID; k += 32) s += bf2f(row[k]) * h[k];
    s = waveReduce(s);
    if (!lane) {
        if (wave < VOC)          outT[wave] = s + bd[wave];
        else if (wave < VOC + 3) rawCS[wave - VOC] = s + bc[wave - VOC];
        else                     rawCS[wave - VOC] = tanhf(s + bsi[wave - VOC - 3]);
    }
}

extern "C" void kernel_launch(void* const* d_in, const int* in_sizes, int n_in,
                              void* d_out, int out_size, void* d_ws, size_t ws_size,
                              hipStream_t stream) {
    const int*   toks   = (const int*)  d_in[0];
    const float* h0     = (const float*)d_in[1];
    const float* c0     = (const float*)d_in[2];
    const float* s0     = (const float*)d_in[3];
    const float* emb    = (const float*)d_in[4];
    const float* Wc     = (const float*)d_in[5];
    const float* bc     = (const float*)d_in[6];
    const float* Wsi    = (const float*)d_in[7];
    const float* bsi    = (const float*)d_in[8];
    const float* WihF   = (const float*)d_in[9];
    const float* WhhF   = (const float*)d_in[10];
    const float* bihF   = (const float*)d_in[11];
    const float* bhhF   = (const float*)d_in[12];
    const float* WihB   = (const float*)d_in[13];
    const float* WhhB   = (const float*)d_in[14];
    const float* bihB   = (const float*)d_in[15];
    const float* bhhB   = (const float*)d_in[16];
    const float* Wd     = (const float*)d_in[17];
    const float* bd     = (const float*)d_in[18];
    float* out = (float*)d_out;

    // ---- workspace layout (all 256B aligned) ----
    uint8_t* ws = (uint8_t*)d_ws;
    size_t o = 0;
    auto take = [&](size_t bytes) { size_t r = o; o += (bytes + 255) & ~size_t(255); return r; };
    unsigned short* WxF  = (unsigned short*)(ws + take((size_t)G4H * KP * 2));
    unsigned short* WxB  = (unsigned short*)(ws + take((size_t)G4H * KP * 2));
    unsigned short* WhFb = (unsigned short*)(ws + take((size_t)G4H * KP * 2));
    unsigned short* WhBb = (unsigned short*)(ws + take((size_t)G4H * KP * 2));
    unsigned short* WsFb = (unsigned short*)(ws + take((size_t)G4H * KPS * 2));
    unsigned short* WsBb = (unsigned short*)(ws + take((size_t)G4H * KPS * 2));
    unsigned short* Eb   = (unsigned short*)(ws + take((size_t)VOC * KP * 2));
    unsigned short* WdB  = (unsigned short*)(ws + take((size_t)VOC * 2 * HID * 2));
    unsigned short* WcB  = (unsigned short*)(ws + take((size_t)3 * 2 * HID * 2));
    unsigned short* WsiB = (unsigned short*)(ws + take((size_t)SW * 2 * HID * 2));
    float*          Xb   = (float*)(ws + take((size_t)2 * TT * G4H * 4));
    float*          gates= (float*)(ws + take((size_t)2 * G4H * 4));
    float*          hS   = (float*)(ws + take((size_t)2 * HID * 4));
    unsigned short* hb   = (unsigned short*)(ws + take((size_t)2 * KP * 2));
    float*          cS   = (float*)(ws + take((size_t)2 * HID * 4));
    float*          stS  = (float*)(ws + take((size_t)SD * SW * 4));
    unsigned short* stb  = (unsigned short*)(ws + take((size_t)KPS * 2));
    float*          rawCS= (float*)(ws + take((size_t)NPROJ * 4));
    float*          trash= (float*)(ws + take((size_t)64 * 4));
    (void)ws_size; (void)in_sizes; (void)n_in; (void)out_size;

    auto gPack = [](long n) { return (int)((n + 255) / 256); };

    // repack weights to bf16 (redone every call -> deterministic)
    kPack<<<gPack((long)G4H * KP),  256, 0, stream>>>(WxF,  WihF, G4H, HID + SW, KP,  0,   HID);
    kPack<<<gPack((long)G4H * KP),  256, 0, stream>>>(WxB,  WihB, G4H, HID + SW, KP,  0,   HID);
    kPack<<<gPack((long)G4H * KPS), 256, 0, stream>>>(WsFb, WihF, G4H, HID + SW, KPS, HID, SW);
    kPack<<<gPack((long)G4H * KPS), 256, 0, stream>>>(WsBb, WihB, G4H, HID + SW, KPS, HID, SW);
    kPack<<<gPack((long)G4H * KP),  256, 0, stream>>>(WhFb, WhhF, G4H, HID, KP, 0, HID);
    kPack<<<gPack((long)G4H * KP),  256, 0, stream>>>(WhBb, WhhB, G4H, HID, KP, 0, HID);
    kPack<<<gPack((long)VOC * KP),  256, 0, stream>>>(Eb,   emb,  VOC, HID, KP, 0, HID);
    kPack<<<gPack((long)VOC * 2 * HID), 256, 0, stream>>>(WdB,  Wd,  VOC, 2 * HID, 2 * HID, 0, 2 * HID);
    kPack<<<gPack((long)3   * 2 * HID), 256, 0, stream>>>(WcB,  Wc,  3,   2 * HID, 2 * HID, 0, 2 * HID);
    kPack<<<gPack((long)SW  * 2 * HID), 256, 0, stream>>>(WsiB, Wsi, SW,  2 * HID, 2 * HID, 0, 2 * HID);

    kInit<<<16, 256, 0, stream>>>(hS, cS, stS, hb, h0, c0, s0);

    // hoisted input-projection GEMM (WMMA): 6000 wave-tiles, 8 waves/block
    kXproj<<<(2 * MT * NT) / 8, 256, 0, stream>>>(WxF, WxB, Eb, toks,
                                                  bihF, bhhF, bihB, bhhB, Xb);

    // seed ctrl/si(0) from h0 (decoder output discarded into trash)
    kProj<<<(NPROJ + 7) / 8, 256, 0, stream>>>(hS, WdB, bd, WcB, bc, WsiB, bsi,
                                               trash, rawCS);

    // sequential scan
    for (int t = 0; t < TT; ++t) {
        kStack<<<1, 512, 0, stream>>>(rawCS, stS, stb);
        kGates<<<2 * MT, 256, 0, stream>>>(WhFb, WhBb, WsFb, WsBb,
                                           hb, stb, Xb, gates, t);
        kUpdate<<<(2 * HID + 255) / 256, 256, 0, stream>>>(gates, cS, hS, hb);
        kProj<<<(NPROJ + 7) / 8, 256, 0, stream>>>(hS, WdB, bd, WcB, bc, WsiB, bsi,
                                                   out + (size_t)t * VOC, rawCS);
    }
}
